// DSMM_6442450944217
// MI455X (gfx1250) — matrix operand order repaired and checked by reading, not compiled, not run on hardware
//
#include <hip/hip_runtime.h>

// CDNA5 / gfx1250, wave32. f16 activations end-to-end, f32 accumulation.
// All channel contractions -> v_wmma_f32_16x16x32_f16.
// Big-N GEMMs: one wave computes a 16x64 strip (4 n-tiles) per K-sweep.

typedef __attribute__((ext_vector_type(16))) _Float16 v16h;
typedef __attribute__((ext_vector_type(8)))  _Float16 v8h;
typedef __attribute__((ext_vector_type(8)))  float    v8f;
typedef __attribute__((ext_vector_type(4)))  float    v4f;

#define BN_EPS 1e-5f
#define WMMA(a, b, c) __builtin_amdgcn_wmma_f32_16x16x32_f16(false, (a), false, (b), (short)0, (c), false, false)

// A-fragment (16x32 f16): per lane-group g the 16 halfs are one
// contiguous K-run of 16: K = k0+16g+0..15.
__device__ __forceinline__ v16h load_a_f16(const _Float16* __restrict__ ap, int k0, int g) {
  v8h lo = *(const v8h*)(ap + k0 + (g << 4));
  v8h hi = *(const v8h*)(ap + k0 + 8 + (g << 4));
  return __builtin_shufflevector(lo, hi, 0, 1, 2, 3, 4, 5, 6, 7, 8, 9, 10, 11, 12, 13, 14, 15);
}

__device__ __forceinline__ v16h load_a_f32(const float* __restrict__ ap, int k0, int g) {
  const v4f* p0 = (const v4f*)(ap + k0 + (g << 4));
  const v4f* p1 = (const v4f*)(ap + k0 + 8 + (g << 4));
  v4f x0 = p0[0], x1 = p0[1], x2 = p1[0], x3 = p1[1];
  v16h a;
  #pragma unroll
  for (int i = 0; i < 4; ++i) {
    a[i]      = (_Float16)x0[i];
    a[4 + i]  = (_Float16)x1[i];
    a[8 + i]  = (_Float16)x2[i];
    a[12 + i] = (_Float16)x3[i];
  }
  return a;
}

// B-fragment when K is contiguous in memory per lane: halfs h=0..15 <-> K=off+h.
__device__ __forceinline__ v16h load_b_contig(const _Float16* __restrict__ p, int off) {
  v8h lo = *(const v8h*)(p + off);
  v8h hi = *(const v8h*)(p + off + 8);
  return __builtin_shufflevector(lo, hi, 0, 1, 2, 3, 4, 5, 6, 7, 8, 9, 10, 11, 12, 13, 14, 15);
}

__device__ __forceinline__ float prelu(float y, float a) {
  return fmaxf(y, 0.f) + a * fminf(y, 0.f);
}

// ---------------------------------------------------------------------------
// VALU stages (bandwidth-bound)
// ---------------------------------------------------------------------------

template <int C, int H, int W>
__global__ void k_multidilate(const float* __restrict__ x, const float* __restrict__ kk,
                              _Float16* __restrict__ out) {
  int idx = blockIdx.x * blockDim.x + threadIdx.x;
  int xw = idx % W; int t = idx / W;
  int y = t % H; t /= H;
  int c = t % C; int b = t / C;
  if (b == 0) { out[idx] = (_Float16)x[idx]; return; }   // out.at[0].set(x[0])
  const float* kp = kk + (size_t)(b * C + c) * 25;
  const float* xp = x + (size_t)(b * C + c) * (H * W);
  float acc = 0.f;
  #pragma unroll
  for (int d = 1; d <= 3; ++d) {
    #pragma unroll
    for (int ky = 0; ky < 5; ++ky) {
      int yy = y + d * (ky - 2);
      if (yy < 0 || yy >= H) continue;
      #pragma unroll
      for (int kx = 0; kx < 5; ++kx) {
        int xx = xw + d * (kx - 2);
        if (xx < 0 || xx >= W) continue;
        acc += kp[ky * 5 + kx] * xp[yy * W + xx];
      }
    }
  }
  out[idx] = (_Float16)acc;
}

template <int H, int W, int OH, int OW>
__global__ void k_bilinear_up(const _Float16* __restrict__ in, _Float16* __restrict__ out) {
  int idx = blockIdx.x * blockDim.x + threadIdx.x;
  int ox = idx % OW; int t = idx / OW;
  int oy = t % OH; t /= OH;                 // t = b*C + c
  const float sy = (float)(H - 1) / (float)(OH - 1);
  const float sx = (float)(W - 1) / (float)(OW - 1);
  float fy = oy * sy, fx = ox * sx;
  int y0 = (int)fy, x0 = (int)fx;
  int y1 = min(y0 + 1, H - 1), x1 = min(x0 + 1, W - 1);
  float wy = fy - (float)y0, wx = fx - (float)x0;
  const _Float16* p = in + (size_t)t * (H * W);
  float a00 = (float)p[y0 * W + x0], a01 = (float)p[y0 * W + x1];
  float a10 = (float)p[y1 * W + x0], a11 = (float)p[y1 * W + x1];
  float r0 = a00 * (1.f - wx) + a01 * wx;
  float r1 = a10 * (1.f - wx) + a11 * wx;
  out[idx] = (_Float16)(r0 * (1.f - wy) + r1 * wy);
}

template <int C, int H, int W>
__global__ void k_cbr_dw3(const _Float16* __restrict__ in, _Float16* __restrict__ out,
                          const float* __restrict__ w,
                          const float* __restrict__ gg, const float* __restrict__ bb,
                          const float* __restrict__ mm, const float* __restrict__ vv,
                          const float* __restrict__ al) {
  int idx = blockIdx.x * blockDim.x + threadIdx.x;
  int x = idx % W; int t = idx / W;
  int y = t % H; t /= H;
  int c = t % C;                            // t = b*C + c
  const _Float16* p = in + (size_t)t * (H * W);
  const float* wp = w + c * 9;
  float acc = 0.f;
  #pragma unroll
  for (int ky = 0; ky < 3; ++ky) {
    int yy = y + ky - 1;
    if (yy < 0 || yy >= H) continue;
    #pragma unroll
    for (int kx = 0; kx < 3; ++kx) {
      int xx = x + kx - 1;
      if (xx < 0 || xx >= W) continue;
      acc += wp[ky * 3 + kx] * (float)p[yy * W + xx];
    }
  }
  float s = gg[c] * rsqrtf(vv[c] + BN_EPS);
  float yv = acc * s + (bb[c] - mm[c] * s);
  out[idx] = (_Float16)prelu(yv, al[c]);
}

__global__ void k_zero(float* __restrict__ p) {
  p[blockIdx.x * blockDim.x + threadIdx.x] = 0.f;
}

// ---------------------------------------------------------------------------
// WMMA stages. One wave = one 16x64 strip (4 n-tiles); grids sized so every
// wave owns a valid strip (EXEC all-ones). Compile-time dims -> imm offsets.
// ---------------------------------------------------------------------------

// 1x1 conv + BN + PReLU: out[b, coOff+co, p] = prelu(bn(sum_ci W[co,ci]*in[b,ci,p]))
template <int Cin, int Cout, int P, typename OutT>
__global__ void k_cbr1x1_wmma(const _Float16* __restrict__ in, OutT* __restrict__ out,
                              const float* __restrict__ Wm,
                              const float* __restrict__ gg, const float* __restrict__ bb,
                              const float* __restrict__ mm, const float* __restrict__ vv,
                              const float* __restrict__ al,
                              int outCtot, int coOff) {
  constexpr int pT = P >> 6, mT = Cout >> 4, per = pT * mT;
  int wave = (blockIdx.x * blockDim.x + threadIdx.x) >> 5;
  int lane = threadIdx.x & 31;
  int b = wave / per, r0 = wave % per;
  int m0 = (r0 / pT) << 4, n0 = (r0 % pT) << 6;
  int g = lane >> 4, ln = lane & 15;
  const float* ap = Wm + (m0 + ln) * Cin;
  const _Float16* bp = in + (size_t)(b * Cin) * P + n0 + ln + (g << 4) * P;
  v8f c0 = {}, c1 = {}, c2 = {}, c3 = {};
  #pragma unroll
  for (int k0 = 0; k0 < Cin; k0 += 32) {
    v16h a = load_a_f32(ap, k0, g);
    v16h b0, b1, b2, b3;
    #pragma unroll
    for (int h = 0; h < 16; ++h) {
      const _Float16* r = bp + (k0 + h) * P;
      b0[h] = r[0]; b1[h] = r[16]; b2[h] = r[32]; b3[h] = r[48];
    }
    c0 = WMMA(a, b0, c0); c1 = WMMA(a, b1, c1);
    c2 = WMMA(a, b2, c2); c3 = WMMA(a, b3, c3);
  }
  int cobase = m0 + (g << 3);
  OutT* po = out + (size_t)(b * outCtot + coOff + cobase) * P + n0 + ln;
  #pragma unroll
  for (int r = 0; r < 8; ++r) {
    float s = gg[cobase + r] * rsqrtf(vv[cobase + r] + BN_EPS);
    float t = bb[cobase + r] - mm[cobase + r] * s;
    float a = al[cobase + r];
    po[r * P]      = (OutT)prelu(c0[r] * s + t, a);
    po[r * P + 16] = (OutT)prelu(c1[r] * s + t, a);
    po[r * P + 32] = (OutT)prelu(c2[r] * s + t, a);
    po[r * P + 48] = (OutT)prelu(c3[r] * s + t, a);
  }
}

// ecorrT[b,d,n] = sum_c linear_e[d,c] * ef[b,c,n]
template <int C, int N>
__global__ void k_ecorr_wmma(const _Float16* __restrict__ ef, const float* __restrict__ le,
                             _Float16* __restrict__ ecT) {
  constexpr int pT = N >> 6, mT = C >> 4, per = pT * mT;
  int wave = (blockIdx.x * blockDim.x + threadIdx.x) >> 5;
  int lane = threadIdx.x & 31;
  int b = wave / per, r0 = wave % per;
  int m0 = (r0 / pT) << 4, n0 = (r0 % pT) << 6;
  int g = lane >> 4, ln = lane & 15;
  const float* ap = le + (m0 + ln) * C;
  const _Float16* bp = ef + (size_t)(b * C) * N + n0 + ln + (g << 4) * N;
  v8f c0 = {}, c1 = {}, c2 = {}, c3 = {};
  #pragma unroll
  for (int k0 = 0; k0 < C; k0 += 32) {
    v16h a = load_a_f32(ap, k0, g);
    v16h b0, b1, b2, b3;
    #pragma unroll
    for (int h = 0; h < 16; ++h) {
      const _Float16* r = bp + (k0 + h) * N;
      b0[h] = r[0]; b1[h] = r[16]; b2[h] = r[32]; b3[h] = r[48];
    }
    c0 = WMMA(a, b0, c0); c1 = WMMA(a, b1, c1);
    c2 = WMMA(a, b2, c2); c3 = WMMA(a, b3, c3);
  }
  _Float16* po = ecT + (size_t)(b * C + m0 + (g << 3)) * N + n0 + ln;
  #pragma unroll
  for (int r = 0; r < 8; ++r) {
    po[r * N]      = (_Float16)c0[r];
    po[r * N + 16] = (_Float16)c1[r];
    po[r * N + 32] = (_Float16)c2[r];
    po[r * N + 48] = (_Float16)c3[r];
  }
}

// Amat[b,i,d] += sum_{n chunk} qf[b,i,n] * ecorrT[b,d,n]   (split-K, f32 atomics)
template <int C, int N, int KSPLIT>
__global__ void k_corrA_wmma(const _Float16* __restrict__ qf, const _Float16* __restrict__ ecT,
                             float* __restrict__ Amat) {
  constexpr int mT = C >> 4, per = mT * mT * KSPLIT, kchunk = N / KSPLIT;
  int wave = (blockIdx.x * blockDim.x + threadIdx.x) >> 5;
  int lane = threadIdx.x & 31;
  int b = wave / per, r0 = wave % per;
  int ks = r0 % KSPLIT; r0 /= KSPLIT;
  int m0 = (r0 / mT) << 4, n0 = (r0 % mT) << 4;
  int g = lane >> 4, ln = lane & 15;
  int kbeg = ks * kchunk, kend = kbeg + kchunk;
  const _Float16* arow = qf + (size_t)(b * C + m0 + ln) * N;   // K contiguous
  const _Float16* brow = ecT + (size_t)(b * C + n0 + ln) * N;  // K contiguous
  v8f c = {};
  for (int k0 = kbeg; k0 < kend; k0 += 32) {
    __builtin_prefetch(arow + k0 + 128, 0, 1);
    __builtin_prefetch(brow + k0 + 128, 0, 1);
    v16h a = load_a_f16(arow, k0, g);
    v16h bf = load_b_contig(brow, k0 + (g << 4));
    c = WMMA(a, bf, c);
  }
  float* po = Amat + (size_t)(b * C + m0 + (g << 3)) * C + n0 + ln;
  #pragma unroll
  for (int r = 0; r < 8; ++r) atomicAdd(po + r * C, c[r]);
}

// A1[b,i,:] = softmax_d(A[b,i,:]);  Bm[b,d,:] = softmax_i(A[b,:,d])   (f16 out)
__global__ void k_softmax(const float* __restrict__ Amat, _Float16* __restrict__ A1,
                          _Float16* __restrict__ Bm, int Bn, int C) {
  int t = blockIdx.x * blockDim.x + threadIdx.x;
  int rows = Bn * C;
  if (t < rows) {
    const float* row = Amat + (size_t)t * C;
    float mx = -3.4e38f;
    for (int d = 0; d < C; ++d) mx = fmaxf(mx, row[d]);
    float s = 0.f;
    for (int d = 0; d < C; ++d) s += __expf(row[d] - mx);
    float inv = 1.f / s;
    _Float16* o = A1 + (size_t)t * C;
    for (int d = 0; d < C; ++d) o[d] = (_Float16)(__expf(row[d] - mx) * inv);
  } else {
    int tt = t - rows;
    int b = tt / C, d = tt % C;
    const float* base = Amat + (size_t)b * C * C + d;   // stride C over i
    float mx = -3.4e38f;
    for (int i = 0; i < C; ++i) mx = fmaxf(mx, base[i * C]);
    float s = 0.f;
    for (int i = 0; i < C; ++i) s += __expf(base[i * C] - mx);
    float inv = 1.f / s;
    _Float16* o = Bm + (size_t)(b * C + d) * C;
    for (int i = 0; i < C; ++i) o[i] = (_Float16)(__expf(base[i * C] - mx) * inv);
  }
}

// dest[b,m,n] = sum_k Asm[b,m,k]*Bbig[b,k,n] + resSrc[b,m,n]
template <int C, int N>
__global__ void k_attmm_wmma(const _Float16* __restrict__ Asm, const _Float16* __restrict__ Bbig,
                             const _Float16* __restrict__ resSrc, _Float16* __restrict__ dest) {
  constexpr int pT = N >> 6, mT = C >> 4, per = pT * mT;
  int wave = (blockIdx.x * blockDim.x + threadIdx.x) >> 5;
  int lane = threadIdx.x & 31;
  int b = wave / per, r0 = wave % per;
  int m0 = (r0 / pT) << 4, n0 = (r0 % pT) << 6;
  int g = lane >> 4, ln = lane & 15;
  const _Float16* ap = Asm + (size_t)(b * C + m0 + ln) * C;  // K contiguous
  const _Float16* bp = Bbig + (size_t)(b * C) * N + n0 + ln + (g << 4) * N;
  v8f c0 = {}, c1 = {}, c2 = {}, c3 = {};
  #pragma unroll
  for (int k0 = 0; k0 < C; k0 += 32) {
    v16h a = load_a_f16(ap, k0, g);
    v16h b0, b1, b2, b3;
    #pragma unroll
    for (int h = 0; h < 16; ++h) {
      const _Float16* r = bp + (k0 + h) * N;
      b0[h] = r[0]; b1[h] = r[16]; b2[h] = r[32]; b3[h] = r[48];
    }
    c0 = WMMA(a, b0, c0); c1 = WMMA(a, b1, c1);
    c2 = WMMA(a, b2, c2); c3 = WMMA(a, b3, c3);
  }
  size_t rb = (size_t)(b * C + m0 + (g << 3)) * N + n0 + ln;
  #pragma unroll
  for (int r = 0; r < 8; ++r) {
    dest[rb + r * N]      = (_Float16)(c0[r] + (float)resSrc[rb + r * N]);
    dest[rb + r * N + 16] = (_Float16)(c1[r] + (float)resSrc[rb + r * N + 16]);
    dest[rb + r * N + 32] = (_Float16)(c2[r] + (float)resSrc[rb + r * N + 32]);
    dest[rb + r * N + 48] = (_Float16)(c3[r] + (float)resSrc[rb + r * N + 48]);
  }
}

// ---------------------------------------------------------------------------

extern "C" void kernel_launch(void* const* d_in, const int* in_sizes, int n_in,
                              void* d_out, int out_size, void* d_ws, size_t ws_size,
                              hipStream_t stream) {
  (void)in_sizes; (void)n_in; (void)out_size; (void)ws_size;
  const int B = 16;
  constexpr int C4 = 96, H4 = 64, W4 = 64, C3 = 32, H3 = 128, W3 = 128;
  constexpr int PIX4 = H4 * W4;   // 4096
  constexpr int N = H3 * W3;      // 16384

#define IN(i) ((const float*)d_in[i])
  const float* x4 = IN(0);  const float* k4 = IN(1);
  const float* x3 = IN(2);  const float* k3 = IN(3);
  // params, depth-first dict order: cbrp = {w, bn{g,b,m,v}, a}
  const float *f4w = IN(4),  *f4g = IN(5),  *f4b = IN(6),  *f4m = IN(7),  *f4v = IN(8),  *f4a = IN(9);
  const float *d4w = IN(10), *d4g = IN(11), *d4b = IN(12), *d4m = IN(13), *d4v = IN(14), *d4a = IN(15);
  const float *p4w = IN(16), *p4g = IN(17), *p4b = IN(18), *p4m = IN(19), *p4v = IN(20), *p4a = IN(21);
  const float *f3w = IN(22), *f3g = IN(23), *f3b = IN(24), *f3m = IN(25), *f3v = IN(26), *f3a = IN(27);
  const float *d3w = IN(28), *d3g = IN(29), *d3b = IN(30), *d3m = IN(31), *d3v = IN(32), *d3a = IN(33);
  const float *p3w = IN(34), *p3g = IN(35), *p3b = IN(36), *p3m = IN(37), *p3v = IN(38), *p3a = IN(39);
  const float *lin = IN(40);
  const float *cdw = IN(41), *cdg = IN(42), *cdb = IN(43), *cdm = IN(44), *cdv = IN(45), *cda = IN(46);
  const float *cpw = IN(47), *cpg = IN(48), *cpb = IN(49), *cpm = IN(50), *cpv = IN(51), *cpa = IN(52);
#undef IN
  float* out = (float*)d_out;

  const size_t n4   = (size_t)B * C4 * H4 * W4;   // 6,291,456
  const size_t n3s  = (size_t)B * C3 * H3 * W3;   // 8,388,608
  const size_t nbig = (size_t)B * C4 * H3 * W3;   // 25,165,824
  const size_t nA   = (size_t)B * C4 * C4;        // 147,456

  char* wsp = (char*)d_ws;
  auto take = [&](size_t bytes) { char* p = wsp; wsp += (bytes + 255) & ~(size_t)255; return p; };
  _Float16* B0 = (_Float16*)take(n3s * 2);   // x4n -> x3all
  _Float16* B1 = (_Float16*)take(n3s * 2);   // x3n
  _Float16* B2 = (_Float16*)take(n3s * 2);   // x4all -> dw3
  _Float16* B3 = (_Float16*)take(nbig * 2);  // x4_up -> query (qf)
  _Float16* B4 = (_Float16*)take(nbig * 2);  // dw4 -> exemplar (ef)
  _Float16* B5 = (_Float16*)take(nbig * 2);  // sum_e
  _Float16* B6 = (_Float16*)take(nbig * 2);  // sum_q
  _Float16* B7 = (_Float16*)take(nbig * 2);  // cc depthwise scratch
  _Float16* H2 = (_Float16*)take(nbig * 2);  // ecorrT
  float*    S0 = (float*)take(nA * 4);       // Amat f32
  _Float16* S1 = (_Float16*)take(nA * 2);    // A1 f16
  _Float16* S2 = (_Float16*)take(nA * 2);    // Bm f16

  const dim3 blk(256);
  const unsigned gBig = (unsigned)(nbig / 256);                 // 98304
  const unsigned gMM  = (unsigned)(B * (C4/16) * (N/64) / 8);   // 3072

  // 1) dyn multi-dilate (f32 in -> f16 out)
  k_multidilate<C4, H4, W4><<<dim3((unsigned)(n4 / 256)), blk, 0, stream>>>(x4, k4, B0);
  k_multidilate<C3, H3, W3><<<dim3((unsigned)(n3s / 256)), blk, 0, stream>>>(x3, k3, B1);

  // 2) fuse4 (1x1, 96->96 @64x64): B0 -> B2
  k_cbr1x1_wmma<C4, C4, PIX4, _Float16><<<dim3((unsigned)(B * 6 * 64 / 8)), blk, 0, stream>>>(
      B0, B2, f4w, f4g, f4b, f4m, f4v, f4a, C4, 0);

  // 3) bilinear x2 upsample: B2 -> B3
  k_bilinear_up<H4, W4, H3, W3><<<dim3(gBig), blk, 0, stream>>>(B2, B3);

  // 4) smooth4_dw: B3 -> B4 ; smooth4_pw: B4 -> B3 (= query)
  k_cbr_dw3<C4, H3, W3><<<dim3(gBig), blk, 0, stream>>>(B3, B4, d4w, d4g, d4b, d4m, d4v, d4a);
  k_cbr1x1_wmma<C4, C4, N, _Float16><<<dim3(gMM), blk, 0, stream>>>(
      B4, B3, p4w, p4g, p4b, p4m, p4v, p4a, C4, 0);

  // 5) fuse3 (1x1, 32->32 @128x128): B1 -> B0
  k_cbr1x1_wmma<C3, C3, N, _Float16><<<dim3((unsigned)(B * 2 * 256 / 8)), blk, 0, stream>>>(
      B1, B0, f3w, f3g, f3b, f3m, f3v, f3a, C3, 0);

  // 6) smooth3_dw: B0 -> B2 ; smooth3_pw (32->96): B2 -> B4 (= exemplar; resize identity)
  k_cbr_dw3<C3, H3, W3><<<dim3((unsigned)(n3s / 256)), blk, 0, stream>>>(B0, B2, d3w, d3g, d3b, d3m, d3v, d3a);
  k_cbr1x1_wmma<C3, C4, N, _Float16><<<dim3(gMM), blk, 0, stream>>>(
      B2, B4, p3w, p3g, p3b, p3m, p3v, p3a, C4, 0);

  // 7) ecorrT[b,d,n] = linear_e (d,c) . ef (c,n)
  k_ecorr_wmma<C4, N><<<dim3(gMM), blk, 0, stream>>>(B4, lin, H2);

  // 8) A[b,i,d] = qf (i,n) . ecorrT (d,n)^T  (split-K over N, atomic f32 accumulate)
  k_zero<<<dim3((unsigned)(nA / 256)), blk, 0, stream>>>(S0);
  k_corrA_wmma<C4, N, 16><<<dim3((unsigned)(B * 6 * 6 * 16 / 8)), blk, 0, stream>>>(B3, H2, S0);

  // 9) softmaxes -> f16 operand matrices
  k_softmax<<<dim3((unsigned)(2 * B * C4 / 256)), blk, 0, stream>>>(S0, S1, S2, B, C4);

  // 10) sum_e = Bm.qf + exemplar -> B5 ; sum_q = A1.ef + query -> B6
  k_attmm_wmma<C4, N><<<dim3(gMM), blk, 0, stream>>>(S2, B3, B4, B5);
  k_attmm_wmma<C4, N><<<dim3(gMM), blk, 0, stream>>>(S1, B4, B3, B6);

  // 11) cc dsconv on e branch -> out channels [0,96)
  k_cbr_dw3<C4, H3, W3><<<dim3(gBig), blk, 0, stream>>>(B5, B7, cdw, cdg, cdb, cdm, cdv, cda);
  k_cbr1x1_wmma<C4, C4, N, float><<<dim3(gMM), blk, 0, stream>>>(
      B7, out, cpw, cpg, cpb, cpm, cpv, cpa, 2 * C4, 0);

  // 12) cc dsconv on q branch -> out channels [96,192)
  k_cbr_dw3<C4, H3, W3><<<dim3(gBig), blk, 0, stream>>>(B6, B7, cdw, cdg, cdb, cdm, cdv, cda);
  k_cbr1x1_wmma<C4, C4, N, float><<<dim3(gMM), blk, 0, stream>>>(
      B7, out, cpw, cpg, cpb, cpm, cpv, cpa, 2 * C4, C4);
}
